// FusedMoE_37752762532030
// MI455X (gfx1250) — matrix operand order repaired and checked by reading, not compiled, not run on hardware
//
#include <hip/hip_runtime.h>

typedef unsigned short u16;
typedef unsigned int   u32;
typedef __attribute__((ext_vector_type(16))) __bf16 v16bf;
typedef __attribute__((ext_vector_type(8)))  float  v8f;

// ---------- constants (problem shape) ----------
#define TOK   1024
#define TOPK  8
#define NEXP  64
#define HDIM  1024
#define IDIM  512            // intermediate
#define I2    1024           // 2*I
#define CAP   256
#define NEXPD 8192           // TOK*TOPK

// ---------- helpers ----------
__device__ __forceinline__ u16 f2bf(float f) {
    u32 u = __float_as_uint(f);
    u32 r = (u + 0x7FFFu + ((u >> 16) & 1u)) >> 16;   // round-to-nearest-even
    return (u16)r;
}
__device__ __forceinline__ u32 pack2(float a, float b) {
    return (u32)f2bf(a) | ((u32)f2bf(b) << 16);
}

union Frag {                 // 32B: one WMMA 16-bit A/B operand per lane
    uint4 u[2];
    v16bf v;
};

// ---- CDNA5 async global->LDS (ASYNCcnt-tracked, no VGPR round trip) ----
// GVS mode: mem_addr = SADDR + VADDR(u32) + IOFFSET ; LDS addr = VDST + IOFFSET.
__device__ __forceinline__ void async_b128(unsigned lds, unsigned goff, const void* base) {
    asm volatile("global_load_async_to_lds_b128 %0, %1, %2"
                 :: "v"(lds), "v"(goff), "s"(base)
                 : "memory");
}
__device__ __forceinline__ void wait_async(void) {           // full drain
    asm volatile("s_wait_asynccnt 0x0" ::: "memory");
}
__device__ __forceinline__ void wait_async_le6(void) {       // oldest tile (6 issues) done
    asm volatile("s_wait_asynccnt 0x6" ::: "memory");
}
__device__ __forceinline__ void wait_async_le4(void) {       // oldest tile (4 issues) done
    asm volatile("s_wait_asynccnt 0x4" ::: "memory");
}
__device__ __forceinline__ unsigned lds_addr_of(const void* p) {
    return (unsigned)(unsigned long long)p;                  // low 32 bits = LDS byte address
}

// ---------- 1) routing: stable per-expert scan ----------
__global__ void moe_route(const int* __restrict__ ids,
                          int* __restrict__ counts,
                          int* __restrict__ starts,
                          int* __restrict__ row_map,
                          int* __restrict__ sort_out) {
    const int e = threadIdx.x;            // blockDim.x == 64
    int cnt = 0;
    for (int i = 0; i < NEXPD; ++i) {
        if (ids[i] == e) {
            if (cnt < CAP) row_map[e * CAP + cnt] = i;
            ++cnt;
        }
    }
    counts[e] = cnt;
    __syncthreads();
    if (e == 0) {
        int s = 0;
        for (int j = 0; j < NEXP; ++j) { starts[j] = s; s += counts[j]; }
    }
    __syncthreads();
    const int s = starts[e];
    const int m = cnt < CAP ? cnt : CAP;
    for (int c = 0; c < m; ++c) sort_out[s + c] = row_map[e * CAP + c];
}

// ---------- 2) weight transpose + f32->bf16 ([e][R][C] -> [e][C][R]) ----------
__global__ void __launch_bounds__(256)
transconv(const float* __restrict__ src, u16* __restrict__ dst, int R, int C) {
    __shared__ float t[32][33];
    const int e = blockIdx.z;
    const float* S = src + (size_t)e * R * C;
    u16* D = dst + (size_t)e * R * C;
    const int c0 = blockIdx.x * 32, r0 = blockIdx.y * 32;
    const int tx = threadIdx.x, ty = threadIdx.y;          // (32,8)
    #pragma unroll
    for (int j = 0; j < 32; j += 8)
        t[ty + j][tx] = S[(size_t)(r0 + ty + j) * C + c0 + tx];
    __syncthreads();
    #pragma unroll
    for (int j = 0; j < 32; j += 8)
        D[(size_t)(c0 + ty + j) * R + r0 + tx] = f2bf(t[tx][ty + j]);
}

// ---------- 3) gather tokens into zero-padded bf16 dispatch buffer ----------
__global__ void __launch_bounds__(128)
moe_gather(const float* __restrict__ hidden,
           const int* __restrict__ row_map,
           const int* __restrict__ counts,
           u16* __restrict__ bufB) {
    const int ec = blockIdx.x;             // e*CAP + c
    const int e = ec >> 8, c = ec & 255;
    int cnt = counts[e]; if (cnt > CAP) cnt = CAP;
    u16* dst = bufB + (size_t)ec * HDIM + threadIdx.x * 8;
    if (c < cnt) {
        const int token = row_map[ec] >> 3;        // /TOPK
        const float* s = hidden + (size_t)token * HDIM + threadIdx.x * 8;
        const float4 f0 = *(const float4*)s;
        const float4 f1 = *(const float4*)(s + 4);
        uint4 p;
        p.x = pack2(f0.x, f0.y); p.y = pack2(f0.z, f0.w);
        p.z = pack2(f1.x, f1.y); p.w = pack2(f1.z, f1.w);
        *(uint4*)dst = p;
    } else {
        *(uint4*)dst = make_uint4(0u, 0u, 0u, 0u);
    }
}

// ---------- 4) grouped GEMM1 (buf @ w13) + SwiGLU -> inter (bf16) ----------
// Double-buffered via async->LDS; all B-frags preloaded so WMMAs issue back-to-back.
__global__ void __launch_bounds__(128)
moe_gemm1(const u16* __restrict__ bufB,    // [E][CAP][H]
          const u16* __restrict__ w13T,    // [E][2I][H]  n-major, k contiguous
          u16* __restrict__ interB) {      // [E][CAP][I]
    const int e  = blockIdx.z;
    const int NB = blockIdx.x * 64;        // gate-column base within I
    const int MB = blockIdx.y * 64;
    const int tid  = threadIdx.x;
    const int lane = tid & 31;
    const int wave = tid >> 5;             // 4 waves: 16 rows each
    const int half = lane >> 4;
    const int l16  = lane & 15;

    __shared__ u16 As[2][64][32];
    __shared__ u16 Bg[2][64][32];
    __shared__ u16 Bu[2][64][32];

    v8f accg[4] = {}, accu[4] = {};

    const u16* Ag = bufB + ((size_t)e * CAP + MB) * HDIM;
    const u16* Wg = w13T + ((size_t)e * I2 + NB) * HDIM;
    const u16* Wu = w13T + ((size_t)e * I2 + IDIM + NB) * HDIM;

    // per-thread chunk coords: 2 x b128 per matrix per tile (idx in [0,256))
    const int r0 = tid >> 2,          q0 = (tid & 3) * 8;
    const int r1 = (tid + 128) >> 2,  q1 = ((tid + 128) & 3) * 8;
    const unsigned lo0 = (unsigned)((r0 * 32 + q0) * 2);
    const unsigned lo1 = (unsigned)((r1 * 32 + q1) * 2);

    // issue one 64x32 tri-tile (A, Bgate, Bup) into LDS buffer p: 6 async issues/thread
    auto issue = [&](int p, int kb) {
        const unsigned a  = lds_addr_of(&As[p][0][0]);
        const unsigned bg = lds_addr_of(&Bg[p][0][0]);
        const unsigned bu = lds_addr_of(&Bu[p][0][0]);
        const unsigned g0 = (unsigned)((r0 * HDIM + kb + q0) * 2);
        const unsigned g1 = (unsigned)((r1 * HDIM + kb + q1) * 2);
        async_b128(a  + lo0, g0, Ag);  async_b128(a  + lo1, g1, Ag);
        async_b128(bg + lo0, g0, Wg);  async_b128(bg + lo1, g1, Wg);
        async_b128(bu + lo0, g0, Wu);  async_b128(bu + lo1, g1, Wu);
    };

    issue(0, 0);
    issue(1, 32);

    int p = 0;
    for (int kb = 0; kb < HDIM; kb += 32, p ^= 1) {
        if (kb + 32 < HDIM) wait_async_le6();   // oldest 6 (tile kb) complete
        else                wait_async();
        __syncthreads();                        // tile p visible to all waves

        // Preload A-frag + all 8 B-frags, then run 8 WMMAs back-to-back.
        Frag a;                                 // A: row = wave*16 + l16
        a.u[0] = *(const uint4*)&As[p][wave * 16 + l16][half * 16];
        a.u[1] = *(const uint4*)&As[p][wave * 16 + l16][half * 16 + 8];

        Frag bg[4], bu[4];
        #pragma unroll
        for (int nb = 0; nb < 4; ++nb) {
            const int n = nb * 16 + l16;
            bg[nb].u[0] = *(const uint4*)&Bg[p][n][half * 16];
            bg[nb].u[1] = *(const uint4*)&Bg[p][n][half * 16 + 8];
            bu[nb].u[0] = *(const uint4*)&Bu[p][n][half * 16];
            bu[nb].u[1] = *(const uint4*)&Bu[p][n][half * 16 + 8];
        }
        #pragma unroll
        for (int nb = 0; nb < 4; ++nb) {
            accg[nb] = __builtin_amdgcn_wmma_f32_16x16x32_bf16(
                false, a.v, false, bg[nb].v, (short)0, accg[nb], false, false);
            accu[nb] = __builtin_amdgcn_wmma_f32_16x16x32_bf16(
                false, a.v, false, bu[nb].v, (short)0, accu[nb], false, false);
        }
        __syncthreads();                        // all waves done reading tile p
        if (kb + 64 < HDIM) issue(p, kb + 64);  // refill p for step k+2
    }

    // SwiGLU fuse + bf16 store; C/D layout: VGPR v -> M = v + 8*half, N = l16
    #pragma unroll
    for (int nb = 0; nb < 4; ++nb) {
        #pragma unroll
        for (int v = 0; v < 8; ++v) {
            const float g = accg[nb][v];
            const float u = accu[nb][v];
            const float sig = 1.0f / (1.0f + __expf(-g));
            const int r = MB + wave * 16 + half * 8 + v;
            const int cc = NB + nb * 16 + l16;
            interB[((size_t)e * CAP + r) * IDIM + cc] = f2bf(g * sig * u);
        }
    }
}

// ---------- 5) grouped GEMM2 (inter @ w2) -> scatter to expanded_out ----------
__global__ void __launch_bounds__(128)
moe_gemm2(const u16* __restrict__ interB,  // [E][CAP][I]
          const u16* __restrict__ w2T,     // [E][H][I]  n-major, k contiguous
          const int* __restrict__ row_map,
          const int* __restrict__ counts,
          float* __restrict__ expanded) {  // [NEXPD][H]
    const int e  = blockIdx.z;
    const int NB = blockIdx.x * 64;        // H-column base
    const int MB = blockIdx.y * 64;
    const int tid  = threadIdx.x;
    const int lane = tid & 31;
    const int wave = tid >> 5;
    const int half = lane >> 4;
    const int l16  = lane & 15;

    __shared__ u16 As[2][64][32];
    __shared__ u16 Bs[2][64][32];

    v8f acc[4] = {};

    const u16* Ap = interB + ((size_t)e * CAP + MB) * IDIM;
    const u16* Wp = w2T + ((size_t)e * HDIM + NB) * IDIM;

    const int r0 = tid >> 2,          q0 = (tid & 3) * 8;
    const int r1 = (tid + 128) >> 2,  q1 = ((tid + 128) & 3) * 8;
    const unsigned lo0 = (unsigned)((r0 * 32 + q0) * 2);
    const unsigned lo1 = (unsigned)((r1 * 32 + q1) * 2);

    auto issue = [&](int p, int kb) {           // 4 async issues/thread
        const unsigned a = lds_addr_of(&As[p][0][0]);
        const unsigned b = lds_addr_of(&Bs[p][0][0]);
        const unsigned g0 = (unsigned)((r0 * IDIM + kb + q0) * 2);
        const unsigned g1 = (unsigned)((r1 * IDIM + kb + q1) * 2);
        async_b128(a + lo0, g0, Ap);  async_b128(a + lo1, g1, Ap);
        async_b128(b + lo0, g0, Wp);  async_b128(b + lo1, g1, Wp);
    };

    issue(0, 0);
    issue(1, 32);

    int p = 0;
    for (int kb = 0; kb < IDIM; kb += 32, p ^= 1) {
        if (kb + 32 < IDIM) wait_async_le4();
        else                wait_async();
        __syncthreads();

        Frag a;
        a.u[0] = *(const uint4*)&As[p][wave * 16 + l16][half * 16];
        a.u[1] = *(const uint4*)&As[p][wave * 16 + l16][half * 16 + 8];

        Frag b[4];
        #pragma unroll
        for (int nb = 0; nb < 4; ++nb) {
            const int n = nb * 16 + l16;
            b[nb].u[0] = *(const uint4*)&Bs[p][n][half * 16];
            b[nb].u[1] = *(const uint4*)&Bs[p][n][half * 16 + 8];
        }
        #pragma unroll
        for (int nb = 0; nb < 4; ++nb) {
            acc[nb] = __builtin_amdgcn_wmma_f32_16x16x32_bf16(
                false, a.v, false, b[nb].v, (short)0, acc[nb], false, false);
        }
        __syncthreads();
        if (kb + 64 < IDIM) issue(p, kb + 64);
    }

    int cnt = counts[e]; if (cnt > CAP) cnt = CAP;
    #pragma unroll
    for (int nb = 0; nb < 4; ++nb) {
        #pragma unroll
        for (int v = 0; v < 8; ++v) {
            const int r = MB + wave * 16 + half * 8 + v;
            if (r < cnt) {
                const int grow = row_map[e * CAP + r];   // expanded row id
                expanded[(size_t)grow * HDIM + NB + nb * 16 + l16] = acc[nb][v];
            }
        }
    }
}

// ---------- launcher ----------
extern "C" void kernel_launch(void* const* d_in, const int* in_sizes, int n_in,
                              void* d_out, int out_size, void* d_ws, size_t ws_size,
                              hipStream_t stream) {
    const float* hidden   = (const float*)d_in[0];   // [1024][1024]
    const float* topk_w   = (const float*)d_in[1];   // [1024][8]
    const int*   topk_ids = (const int*)d_in[2];     // [1024][8]
    const float* w13      = (const float*)d_in[3];   // [64][1024][1024]
    const float* w2       = (const float*)d_in[4];   // [64][512][1024]

    float* out        = (float*)d_out;
    float* out_hidden = out;                                  // 1,048,576
    float* out_exp    = out + 1048576;                        // 8,388,608
    float* out_tw     = out + 1048576 + 8388608;              // 8,192
    int*   out_idx    = (int*)(out + 1048576 + 8388608 + 8192);

    char* ws = (char*)d_ws;
    int* counts   = (int*)(ws + 0);
    int* starts   = (int*)(ws + 256);
    int* row_map  = (int*)(ws + 512);                 // 64*256*4 = 65536
    u16* bufB     = (u16*)(ws + 66048);               // 33,554,432
    u16* interB   = (u16*)(ws + 33620480);            // 16,777,216
    u16* w13T     = (u16*)(ws + 50397696);            // 134,217,728
    u16* w2T      = (u16*)(ws + 184615424);           // 67,108,864  (end 251,724,288)

    // pass-through outputs
    hipMemcpyAsync(out_hidden, hidden, (size_t)1048576 * 4, hipMemcpyDeviceToDevice, stream);
    hipMemcpyAsync(out_tw, topk_w, (size_t)8192 * 4, hipMemcpyDeviceToDevice, stream);

    moe_route<<<1, 64, 0, stream>>>(topk_ids, counts, starts, row_map, out_idx);
    transconv<<<dim3(32, 32, NEXP), dim3(32, 8), 0, stream>>>(w13, w13T, HDIM, I2);
    transconv<<<dim3(32, 16, NEXP), dim3(32, 8), 0, stream>>>(w2, w2T, IDIM, HDIM);
    moe_gather<<<NEXP * CAP, 128, 0, stream>>>(hidden, row_map, counts, bufB);
    moe_gemm1<<<dim3(IDIM / 64, CAP / 64, NEXP), 128, 0, stream>>>(bufB, w13T, interB);
    moe_gemm2<<<dim3(HDIM / 64, CAP / 64, NEXP), 128, 0, stream>>>(interB, w2T, row_map, counts, out_exp);
}
